// MultiHeadAttention_25924422599062
// MI455X (gfx1250) — matrix operand order repaired and checked by reading, not compiled, not run on hardware
//
#include <hip/hip_runtime.h>

// ---------------------------------------------------------------------------
// MultiHeadAttention forward for MI455X (gfx1250)
//   - WMMA f16 (f32 accumulate) for all GEMMs
//   - flash attention with online softmax (exp2 domain)
//   - Tensor Data Mover (tensor_load_to_lds) double-buffered K/V tile DMA
// B=2, S=2048, E=1024, H=16, D=64
// ---------------------------------------------------------------------------

#define EMBED 1024
#define HEADS 16
#define HDIM  64
#define BATCH 2
#define SEQ   2048
#define MROWS (BATCH * SEQ)     // 4096

typedef __attribute__((ext_vector_type(16))) _Float16 v16h;
typedef __attribute__((ext_vector_type(8)))  float    v8f;
typedef __attribute__((ext_vector_type(4)))  unsigned v4u;
typedef __attribute__((ext_vector_type(8)))  int      v8i;
typedef __attribute__((ext_vector_type(4)))  int      v4i;

union FragH {                    // 16 halves = 8 VGPRs, viewed as 2 x uint4
    v16h  h;
    uint4 q[2];
};

__device__ __forceinline__ v8f wmma_f16(const FragH& a, const FragH& b, v8f c) {
    return __builtin_amdgcn_wmma_f32_16x16x32_f16(
        /*neg_a=*/false, a.h, /*neg_b=*/false, b.h,
        /*c_mod=*/(short)0, c, /*reuse_a=*/false, /*reuse_b=*/false);
}

// Butterfly ops within 16-lane halves via ds_swizzle (xor mode, and_mask=0x1f)
template <int MASK>
__device__ __forceinline__ float swz_xor(float v) {
    return __int_as_float(
        __builtin_amdgcn_ds_swizzle(__float_as_int(v), (MASK << 10) | 0x1f));
}
__device__ __forceinline__ float red_max16(float v) {
    v = fmaxf(v, swz_xor<1>(v));
    v = fmaxf(v, swz_xor<2>(v));
    v = fmaxf(v, swz_xor<4>(v));
    v = fmaxf(v, swz_xor<8>(v));
    return v;
}
__device__ __forceinline__ float red_sum16(float v) {
    v += swz_xor<1>(v);
    v += swz_xor<2>(v);
    v += swz_xor<4>(v);
    v += swz_xor<8>(v);
    return v;
}

__device__ __forceinline__ unsigned lds_offset(const void* p) {
    return (unsigned)(uintptr_t)p;   // generic LDS address: low 32 bits = offset
}

// ---------------------------------------------------------------------------
// TDM: load a 2-D f16 tile (tile_w halves x tile_h rows, row stride in halves)
// into LDS, padding each 64-half row by 4 DWORDs (-> 72-half LDS stride).
// D# built per CDNA5 ISA. Tracked by per-wave TENSORcnt.
// ---------------------------------------------------------------------------
__device__ __forceinline__ void tdm_load_tile_f16(unsigned lds_off,
                                                  const _Float16* gsrc,
                                                  unsigned tile_w, unsigned tile_h,
                                                  unsigned long stride_halves) {
    const unsigned long ga = (unsigned long)(uintptr_t)gsrc;
    v4u g0;
    g0[0] = 1u;                                    // count=1, user mode, no gather
    g0[1] = lds_off;                               // lds_addr (bytes)
    g0[2] = (unsigned)ga;                          // global_addr[31:0]
    g0[3] = (unsigned)((ga >> 32) & 0x01ffffffu)   // global_addr[56:32]
            | 0x80000000u;                         // type=2 (bits 127:126 = 10b)
    v8i g1;
    g1[0] = (int)((1u << 16)        // data_size = 2 bytes
                  | (1u << 20)      // pad_enable
                  | (4u << 22)      // pad_interval: 32 DWORDs (= 64 halves/row)
                  | (3u << 25));    // pad_amount: 4 DWORDs (= 8 halves)
    g1[1] = (int)0xffff0000u;                      // tensor_dim0 = 0x7fffffff (lo16)
    g1[2] = (int)0xffff7fffu;                      // dim1[15:0] | dim0[31:16]
    g1[3] = (int)((tile_w << 16) | 0x7fffu);       // tile_dim0 | dim1[31:16]
    g1[4] = (int)(tile_h & 0xffffu);               // tile_dim1 (tile_dim2 = 0)
    g1[5] = (int)(unsigned)(stride_halves & 0xffffffffu);   // dim0_stride lo
    g1[6] = (int)(unsigned)((stride_halves >> 32) & 0xffffu);
    g1[7] = 0;
    v4i g2 = {0, 0, 0, 0};
    v4i g3 = {0, 0, 0, 0};
#if defined(__clang_major__) && (__clang_major__ >= 23)
    v8i g4 = {0, 0, 0, 0, 0, 0, 0, 0};
    __builtin_amdgcn_tensor_load_to_lds(g0, g1, g2, g3, g4, 0);
#else
    __builtin_amdgcn_tensor_load_to_lds(g0, g1, g2, g3, 0);
#endif
}

// ---------------------------------------------------------------------------
// GEMM: out = (A[M,K] @ W[N,K]^T + bias) * alpha.  256 thr (8 waves),
// 128x128 tile, BK=32; wave = 32x64 (8 WMMA).  Register-prefetch staging.
// vt!=0: scatter-store output in V-transposed layout [b,h,d,s] (f16 only).
// ---------------------------------------------------------------------------
#define LDT 40   // LDS row stride in halves (80B, 16B-aligned)

template <typename T> struct StageReg;
template <> struct StageReg<float>    { float4 r[4]; };
template <> struct StageReg<_Float16> { uint4  r[2]; };

__device__ __forceinline__ void stage_fetch(const float* __restrict__ src,
                                            StageReg<float>& g) {
    const float4* s = reinterpret_cast<const float4*>(src);
#pragma unroll
    for (int i = 0; i < 4; ++i) g.r[i] = s[i];
}
__device__ __forceinline__ void stage_fetch(const _Float16* __restrict__ src,
                                            StageReg<_Float16>& g) {
    const uint4* s = reinterpret_cast<const uint4*>(src);
    g.r[0] = s[0];
    g.r[1] = s[1];
}
__device__ __forceinline__ void stage_store(const StageReg<float>& g, _Float16* dst) {
#pragma unroll
    for (int i = 0; i < 4; ++i) {
        dst[4 * i + 0] = (_Float16)g.r[i].x;
        dst[4 * i + 1] = (_Float16)g.r[i].y;
        dst[4 * i + 2] = (_Float16)g.r[i].z;
        dst[4 * i + 3] = (_Float16)g.r[i].w;
    }
}
__device__ __forceinline__ void stage_store(const StageReg<_Float16>& g, _Float16* dst) {
    reinterpret_cast<uint4*>(dst)[0] = g.r[0];
    reinterpret_cast<uint4*>(dst)[1] = g.r[1];
}

template <typename AT, typename OT>
__global__ __launch_bounds__(256) void gemm_wt_bias_kernel(
    const AT* __restrict__ A, const float* __restrict__ W,
    const float* __restrict__ bias, OT* __restrict__ out,
    int M, int N, int K, float alpha, int vt) {
    __shared__ __align__(16) _Float16 sA[128 * LDT];
    __shared__ __align__(16) _Float16 sB[128 * LDT];

    const int t    = threadIdx.x;
    const int lane = t & 31;
    const int w    = t >> 5;
    const int wm   = w >> 1;
    const int wn   = w & 1;
    const int lm   = lane & 15;
    const int kh   = lane >> 4;
    const int m0   = blockIdx.y * 128;
    const int n0   = blockIdx.x * 128;

    const int srow = t >> 1;
    const int scol = (t & 1) * 16;
    const AT*    Ar = A + (size_t)(m0 + srow) * K + scol;
    const float* Wr = W + (size_t)(n0 + srow) * K + scol;

    v8f c[2][4];
#pragma unroll
    for (int i = 0; i < 2; ++i)
#pragma unroll
        for (int j = 0; j < 4; ++j)
#pragma unroll
            for (int e = 0; e < 8; ++e) c[i][j][e] = 0.0f;

    StageReg<AT>    ra;
    StageReg<float> rw;
    stage_fetch(Ar, ra);
    stage_fetch(Wr, rw);

    for (int kt = 0; kt < K; kt += 32) {
        stage_store(ra, &sA[srow * LDT + scol]);
        stage_store(rw, &sB[srow * LDT + scol]);
        __syncthreads();
        if (kt + 32 < K) {                 // prefetch next K-slab during WMMAs
            stage_fetch(Ar + kt + 32, ra);
            stage_fetch(Wr + kt + 32, rw);
        }

        FragH a[2], b[4];
#pragma unroll
        for (int mt = 0; mt < 2; ++mt) {
            const uint4* rb = (const uint4*)&sA[(wm * 32 + mt * 16 + lm) * LDT];
            a[mt].q[0] = rb[kh];
            a[mt].q[1] = rb[2 + kh];
        }
#pragma unroll
        for (int nt = 0; nt < 4; ++nt) {
            const uint4* rb = (const uint4*)&sB[(wn * 64 + nt * 16 + lm) * LDT];
            b[nt].q[0] = rb[kh];
            b[nt].q[1] = rb[2 + kh];
        }
#pragma unroll
        for (int mt = 0; mt < 2; ++mt)
#pragma unroll
            for (int nt = 0; nt < 4; ++nt)
                c[mt][nt] = wmma_f16(a[mt], b[nt], c[mt][nt]);
        __syncthreads();
    }

#pragma unroll
    for (int mt = 0; mt < 2; ++mt)
#pragma unroll
        for (int nt = 0; nt < 4; ++nt) {
            const int col = n0 + wn * 64 + nt * 16 + lm;
            const float bv = bias[col];
#pragma unroll
            for (int vi = 0; vi < 8; ++vi) {
                const int   row = m0 + wm * 32 + mt * 16 + kh * 8 + vi;
                const float val = (c[mt][nt][vi] + bv) * alpha;
                if (vt) {   // [b,h,d,s] layout so TDM can tile V directly
                    const int bb = row >> 11;            // / SEQ
                    const int ss = row & (SEQ - 1);
                    const size_t idx =
                        (((size_t)bb * HEADS + (col >> 6)) * HDIM + (col & (HDIM - 1)))
                            * SEQ + ss;
                    out[idx] = (OT)val;
                } else {
                    out[(size_t)row * N + col] = (OT)val;
                }
            }
        }
}

// ---------------------------------------------------------------------------
// Flash attention.  Block = (b,h) x 128 q-rows, 8 waves x 16 rows.
// K/V 64-key tiles DMA'd by the TDM (wave0: K, wave1: V), double buffered.
// Q pre-scaled by log2e/sqrt(E) so probabilities are exp2(s - rowmax).
// ---------------------------------------------------------------------------
#define LKV 72   // LDS row stride in halves (144B, 16B-aligned) == TDM pad

__global__ __launch_bounds__(256) void flash_attn_kernel(
    const _Float16* __restrict__ Qm, const _Float16* __restrict__ Km,
    const _Float16* __restrict__ Vt, _Float16* __restrict__ Om) {
    __shared__ __align__(16) _Float16 sK[2][64 * LKV];     // [key][d]
    __shared__ __align__(16) _Float16 sV[2][64 * LKV];     // [d][key]
    __shared__ __align__(16) _Float16 sP[8 * 16 * LKV];    // per-wave [row][key]

    const int t    = threadIdx.x;
    const int lane = t & 31;
    const int w    = t >> 5;
    const int lm   = lane & 15;
    const int kh   = lane >> 4;
    const int bh   = blockIdx.y;
    const int b    = bh / HEADS;
    const int h    = bh % HEADS;
    const int q0   = blockIdx.x * 128 + w * 16;

    const _Float16* Kh = Km + (size_t)b * SEQ * EMBED + h * HDIM;
    const _Float16* Vh = Vt + (size_t)bh * HDIM * SEQ;

    // ---- Q fragments (16 rows x 64 d), reused across all KV tiles ----
    const size_t baseQ = ((size_t)b * SEQ + q0) * EMBED + h * HDIM;
    FragH aq[2];
#pragma unroll
    for (int j = 0; j < 2; ++j) {
        const uint4* rb = (const uint4*)(Qm + baseQ + (size_t)lm * EMBED + j * 32);
        aq[j].q[0] = rb[kh];
        aq[j].q[1] = rb[2 + kh];
    }

    v8f   o[4];
    float mstate[8], lstate[8];
#pragma unroll
    for (int dt = 0; dt < 4; ++dt)
#pragma unroll
        for (int e = 0; e < 8; ++e) o[dt][e] = 0.0f;
#pragma unroll
    for (int e = 0; e < 8; ++e) { mstate[e] = -1.0e30f; lstate[e] = 0.0f; }

    _Float16* pw = &sP[w * 16 * LKV];

    // prologue: DMA tile 0 into buffer 0
    if (w == 0)
        tdm_load_tile_f16(lds_offset(&sK[0][0]), Kh, 64, 64, EMBED);
    else if (w == 1)
        tdm_load_tile_f16(lds_offset(&sV[0][0]), Vh, 64, 64, SEQ);

    for (int kb = 0; kb < SEQ; kb += 64) {
        const int buf = (kb >> 6) & 1;
        const int kn  = kb + 64;
        __syncthreads();   // everyone done reading buf^1 from iteration kb-64
        if (w == 0) {
            if (kn < SEQ) {
                tdm_load_tile_f16(lds_offset(&sK[buf ^ 1][0]),
                                  Kh + (size_t)kn * EMBED, 64, 64, EMBED);
                __builtin_amdgcn_s_wait_tensorcnt(1);   // tile kb done
            } else {
                __builtin_amdgcn_s_wait_tensorcnt(0);
            }
        } else if (w == 1) {
            if (kn < SEQ) {
                tdm_load_tile_f16(lds_offset(&sV[buf ^ 1][0]),
                                  Vh + kn, 64, 64, SEQ);
                __builtin_amdgcn_s_wait_tensorcnt(1);
            } else {
                __builtin_amdgcn_s_wait_tensorcnt(0);
            }
        }
        __syncthreads();   // tile kb visible to all waves

        // ---- scores: 16 q-rows x 64 keys ----
        v8f s[4];
#pragma unroll
        for (int nt = 0; nt < 4; ++nt) {
            const uint4* rb = (const uint4*)&sK[buf][(nt * 16 + lm) * LKV];
            FragH bk0, bk1;
            bk0.q[0] = rb[kh];
            bk0.q[1] = rb[2 + kh];
            bk1.q[0] = rb[4 + kh];
            bk1.q[1] = rb[6 + kh];
            v8f z;
#pragma unroll
            for (int e = 0; e < 8; ++e) z[e] = 0.0f;
            s[nt] = wmma_f16(aq[0], bk0, z);
            s[nt] = wmma_f16(aq[1], bk1, s[nt]);
        }

        // ---- online softmax (row = 8*kh + vi, col = lane&15 in half) ----
#pragma unroll
        for (int vi = 0; vi < 8; ++vi) {
            float rm = red_max16(
                fmaxf(fmaxf(s[0][vi], s[1][vi]), fmaxf(s[2][vi], s[3][vi])));
            const float mnew = fmaxf(mstate[vi], rm);
            const float scl  = __builtin_amdgcn_exp2f(mstate[vi] - mnew);
            mstate[vi] = mnew;
            float rs = 0.0f;
#pragma unroll
            for (int nt = 0; nt < 4; ++nt) {
                float p = __builtin_amdgcn_exp2f(s[nt][vi] - mnew);
                s[nt][vi] = p;
                rs += p;
            }
            rs = red_sum16(rs);
            lstate[vi] = lstate[vi] * scl + rs;
#pragma unroll
            for (int dt = 0; dt < 4; ++dt) o[dt][vi] *= scl;
        }

        // ---- P (f16) to per-wave LDS scratch [row][key] ----
#pragma unroll
        for (int nt = 0; nt < 4; ++nt)
#pragma unroll
            for (int vi = 0; vi < 8; ++vi)
                pw[(kh * 8 + vi) * LKV + nt * 16 + lm] = (_Float16)s[nt][vi];

        // ---- O += P @ V ----
#pragma unroll
        for (int j = 0; j < 2; ++j) {
            FragH ap;
            const uint4* rb = (const uint4*)&pw[lm * LKV + j * 32];
            ap.q[0] = rb[kh];
            ap.q[1] = rb[2 + kh];
#pragma unroll
            for (int dt = 0; dt < 4; ++dt) {
                FragH bv;
                const uint4* vb =
                    (const uint4*)&sV[buf][(dt * 16 + lm) * LKV + j * 32];
                bv.q[0] = vb[kh];
                bv.q[1] = vb[2 + kh];
                o[dt] = wmma_f16(ap, bv, o[dt]);
            }
        }
    }

    // ---- normalize and store f16 ----
    const size_t baseO = ((size_t)b * SEQ + q0) * EMBED + h * HDIM;
#pragma unroll
    for (int vi = 0; vi < 8; ++vi) {
        const float rl = __builtin_amdgcn_rcpf(lstate[vi]);
#pragma unroll
        for (int dt = 0; dt < 4; ++dt)
            Om[baseO + (size_t)(kh * 8 + vi) * EMBED + dt * 16 + lm] =
                (_Float16)(o[dt][vi] * rl);
    }
}

// ---------------------------------------------------------------------------
extern "C" void kernel_launch(void* const* d_in, const int* in_sizes, int n_in,
                              void* d_out, int out_size, void* d_ws,
                              size_t ws_size, hipStream_t stream) {
    (void)in_sizes; (void)n_in; (void)out_size; (void)ws_size;
    const float* query = (const float*)d_in[0];
    const float* key   = (const float*)d_in[1];
    const float* value = (const float*)d_in[2];
    const float* Wq    = (const float*)d_in[3];
    const float* bq    = (const float*)d_in[4];
    const float* Wk    = (const float*)d_in[5];
    const float* bk    = (const float*)d_in[6];
    const float* Wv    = (const float*)d_in[7];
    const float* bv    = (const float*)d_in[8];
    const float* Wo    = (const float*)d_in[9];
    const float* bo    = (const float*)d_in[10];

    const size_t MAT = (size_t)MROWS * EMBED;
    _Float16* Q16 = (_Float16*)d_ws;
    _Float16* K16 = Q16 + MAT;
    _Float16* V16 = K16 + MAT;       // [b,h,d,s] transposed layout
    _Float16* A16 = V16 + MAT;

    dim3 blk(256);
    dim3 gProj(EMBED / 128, MROWS / 128);   // 8 x 32
    dim3 gAttn(SEQ / 128, BATCH * HEADS);   // 16 x 32

    const float alphaQ = 1.4426950408889634f / 32.0f;  // log2e / sqrt(E)

    gemm_wt_bias_kernel<float, _Float16><<<gProj, blk, 0, stream>>>(
        query, Wq, bq, Q16, MROWS, EMBED, EMBED, alphaQ, 0);
    gemm_wt_bias_kernel<float, _Float16><<<gProj, blk, 0, stream>>>(
        key, Wk, bk, K16, MROWS, EMBED, EMBED, 1.0f, 0);
    gemm_wt_bias_kernel<float, _Float16><<<gProj, blk, 0, stream>>>(
        value, Wv, bv, V16, MROWS, EMBED, EMBED, 1.0f, 1);

    flash_attn_kernel<<<gAttn, blk, 0, stream>>>(Q16, K16, V16, A16);

    gemm_wt_bias_kernel<_Float16, float><<<gProj, blk, 0, stream>>>(
        A16, Wo, bo, (float*)d_out, MROWS, EMBED, EMBED, 1.0f, 0);
}
